// ContrastDrivenFeatureAggregation_12214886990438
// MI455X (gfx1250) — matrix operand order repaired and checked by reading, not compiled, not run on hardware
//
#include <hip/hip_runtime.h>
#include <hip/hip_bf16.h>

typedef __attribute__((ext_vector_type(16))) __bf16 v16bf;
typedef __attribute__((ext_vector_type(8)))  float  v8f;

#define BATCH 4
#define CCH   256
#define HH    64
#define WWID  64
#define LPIX  4096          // HH*WWID
#define NATT  648           // HEADS * 81
#define ATT_SCALE 0.17677669529663689f   // 32^-0.5

// K-bit permutation so each half-lane's 16 fragment elements are contiguous:
// A-frag element i (half h) needs K = (i&7) + 8*h + ((i&8)<<1)  ==  inv_perm(h*16+i)
__device__ __forceinline__ int kperm(int k) {
    return (k & 7) | ((k & 16) >> 1) | ((k & 8) << 1);
}

// ---------------------------------------------------------------------------
// Weight pack kernels: fp32 -> bf16 in exact WMMA B-fragment lane order.
// Layout: pk[((kt*ntilesPad + nt)*32 + lane)*16 + i],  i: K = kt*32 + half*16 + i,
// col = nt*16 + (lane&15).  Zero-fill beyond N.
// ---------------------------------------------------------------------------
__global__ __launch_bounds__(256)
void pack_w_gemm(const float* __restrict__ W, __bf16* __restrict__ out,
                 int K, int N, int ntilesPad)
{
    int gid = blockIdx.x * 256 + threadIdx.x;
    int total = (K / 32) * ntilesPad * 32;
    if (gid >= total) return;
    int lane = gid & 31;
    int t    = gid >> 5;
    int nt   = t % ntilesPad;
    int kt   = t / ntilesPad;
    int half = lane >> 4, lm = lane & 15;
    int col  = nt * 16 + lm;
    v16bf v;
    #pragma unroll
    for (int i = 0; i < 16; ++i) {
        int kk = kt * 32 + half * 16 + i;
        v[i] = (col < N) ? (__bf16)W[(size_t)kk * N + col] : (__bf16)0.f;
    }
    *reinterpret_cast<v16bf*>(out + (size_t)gid * 16) = v;
}

// Conv weights (O=256, I=256, 3, 3), K ordered (cg, tap): ktIdx = cg*9 + tap,
// fragment K = ci offset within the 32-ci group.
__global__ __launch_bounds__(256)
void pack_w_conv(const float* __restrict__ wgt, __bf16* __restrict__ out)
{
    int gid = blockIdx.x * 256 + threadIdx.x;      // 72*16*32 = 36864 threads
    if (gid >= 72 * 16 * 32) return;
    int lane = gid & 31;
    int t    = gid >> 5;                            // ktIdx*16 + oct
    int oct  = t & 15;
    int ktIdx = t >> 4;                             // cg*9 + tap
    int cg = ktIdx / 9, tap = ktIdx % 9;
    int half = lane >> 4, lm = lane & 15;
    int oc = oct * 16 + lm;
    v16bf v;
    #pragma unroll
    for (int i = 0; i < 16; ++i)
        v[i] = (__bf16)wgt[(size_t)oc * 2304 + (size_t)(cg * 32 + half * 16 + i) * 9 + tap];
    *reinterpret_cast<v16bf*>(out + (size_t)gid * 16) = v;
}

// ---------------------------------------------------------------------------
// Conv 3x3 (pad 1) + scale/bias + SiLU, NCHW in/out.  Implicit GEMM, tap-major K.
// Block = 32 pixels x 256 oc.  8 waves = 2 pixel-subtiles x 4 oc-groups; each
// wave = 16 pixels x 64 oc (4 accumulators).  Input slab (32ci x 3 x 34) in LDS;
// one A-fragment build feeds 4 WMMAs.  B frags: one v16bf load each (packed bf16).
// ---------------------------------------------------------------------------
__global__ __launch_bounds__(256)
void conv3x3_bn_silu(const float* __restrict__ in, const __bf16* __restrict__ wpk,
                     const float* __restrict__ scale, const float* __restrict__ bias,
                     float* __restrict__ out)
{
    __shared__ float slab[32][3][34];
    const int tid  = threadIdx.x;
    const int lane = tid & 31;
    const int wv   = tid >> 5;                 // wave 0..7
    const int mwv  = wv & 1;                   // pixel subtile
    const int octBase = (wv >> 1) * 4;         // 4 oc-tiles per wave
    const int bx   = blockIdx.x;               // 0..511  (b, y, xTile32)
    const int x0   = (bx & 1) * 32;
    const int y    = (bx >> 1) & 63;
    const int b    = bx >> 7;
    const int half = lane >> 4;
    const int lm   = lane & 15;
    const int pcol = mwv * 16 + lm;            // pixel column within 34-wide slab
    const v16bf* wp = reinterpret_cast<const v16bf*>(wpk);

    v8f acc[4] = {};
    for (int cg = 0; cg < 8; ++cg) {
        const int c0 = cg * 32;
        for (int e = tid; e < 32 * 3 * 34; e += 256) {     // zero-padded slab
            int ci = e / 102, rem = e % 102;
            int rr = rem / 34, cc = rem % 34;
            int yy = y - 1 + rr, xx = x0 - 1 + cc;
            float v = 0.f;
            if (yy >= 0 && yy < HH && xx >= 0 && xx < WWID)
                v = in[(((b * CCH) + c0 + ci) * HH + yy) * WWID + xx];
            slab[ci][rr][cc] = v;
        }
        __syncthreads();
        #pragma unroll
        for (int tap = 0; tap < 9; ++tap) {
            const int ti = tap / 3, tj = tap % 3;
            v16bf af;
            #pragma unroll
            for (int i = 0; i < 16; ++i) {
                int kc = half * 16 + i;
                af[i] = (__bf16)slab[kc][ti][pcol + tj];
            }
            const size_t kb = ((size_t)(cg * 9 + tap) * 16 + octBase) * 32 + lane;
            #pragma unroll
            for (int j = 0; j < 4; ++j) {
                v16bf bf = wp[kb + (size_t)j * 32];
                acc[j] = __builtin_amdgcn_wmma_f32_16x16x32_bf16(
                             false, af, false, bf, (short)0, acc[j], false, false);
            }
        }
        __syncthreads();
    }
    #pragma unroll
    for (int j = 0; j < 4; ++j) {
        const int oc = (octBase + j) * 16 + lm;
        const float s = scale[oc], bb = bias[oc];
        const size_t rowBase = (((size_t)b * CCH + oc) * HH + y) * WWID;
        #pragma unroll
        for (int r = 0; r < 8; ++r) {
            int xp = x0 + mwv * 16 + r + 8 * half;
            float v = acc[j][r] * s + bb;
            out[rowBase + xp] = v / (1.f + __expf(-v));   // SiLU
        }
    }
}

// ---------------------------------------------------------------------------
// GEMM: out[M x N] = A[M x K] (pixel-major fp32) @ W + bias, bf16 WMMA.
// Block tile 64 x 128 (8 waves: 4 M x 2 N; wave = 16 rows x 64 cols, 4 accs).
// A chunk staged in LDS as bf16 row-major in natural k order -> one v16bf LDS
// load per wave per k-step.  B from pre-packed bf16 fragments.
// outNCHW re-transposes rows into NCHW.
// ---------------------------------------------------------------------------
__global__ __launch_bounds__(256)
void gemm_bias_wmma(const float* __restrict__ A, const __bf16* __restrict__ Wpk,
                    const float* __restrict__ bias, float* __restrict__ out,
                    int K, int N, int ntilesPad, int outNCHW)
{
    __shared__ alignas(32) __bf16 At[64 * 32];
    const int tid = threadIdx.x, lane = tid & 31, wv = tid >> 5;
    const int mw = wv >> 1, nw = wv & 1;
    const int m0 = blockIdx.x * 64;
    const int half = lane >> 4, lm = lane & 15;
    const int mrow = mw * 16 + lm;
    const int ntBase = blockIdx.y * 8 + nw * 4;
    const int n0 = blockIdx.y * 128 + nw * 64;
    const v16bf* wp = reinterpret_cast<const v16bf*>(Wpk);

    v8f acc[4] = {};
    for (int kt = 0; kt < K / 32; ++kt) {
        const int k0 = kt * 32;
        if (k0 + 32 < K)
            __builtin_prefetch(&A[(size_t)(m0 + (tid >> 2)) * K + k0 + 32], 0, 1);
        #pragma unroll
        for (int e = 0; e < 8; ++e) {
            int idx = tid + e * 256;
            int row = idx >> 5, k = idx & 31;
            At[row * 32 + k] = (__bf16)A[(size_t)(m0 + row) * K + k0 + k];
        }
        __syncthreads();
        v16bf af = *reinterpret_cast<const v16bf*>(At + mrow * 32 + half * 16);
        const size_t kb = ((size_t)kt * ntilesPad + ntBase) * 32 + lane;
        #pragma unroll
        for (int j = 0; j < 4; ++j) {
            v16bf bf = wp[kb + (size_t)j * 32];
            acc[j] = __builtin_amdgcn_wmma_f32_16x16x32_bf16(
                         false, af, false, bf, (short)0, acc[j], false, false);
        }
        __syncthreads();
    }
    #pragma unroll
    for (int j = 0; j < 4; ++j) {
        const int col = n0 + j * 16 + lm;
        if (col >= N) continue;
        const float bv = bias[col];
        #pragma unroll
        for (int r = 0; r < 8; ++r) {
            int m = m0 + mw * 16 + r + 8 * half;      // global row = b*LPIX + l
            float v = acc[j][r] + bv;
            if (outNCHW) {
                int b = m >> 12, l = m & (LPIX - 1);
                out[(((size_t)b * CCH + col) << 12) + l] = v;
            } else {
                out[(size_t)m * N + col] = v;
            }
        }
    }
}

// ---------------------------------------------------------------------------
// Haar split + NCHW -> pixel-major transpose.
// ---------------------------------------------------------------------------
__global__ __launch_bounds__(256)
void haar_transpose(const float* __restrict__ t2, float* __restrict__ xc,
                    float* __restrict__ fg, float* __restrict__ bg)
{
    int g = blockIdx.x * 256 + threadIdx.x;     // over B*L*C, c fastest
    int c = g & (CCH - 1);
    int l = (g >> 8) & (LPIX - 1);
    int b = g >> 20;
    int y = l >> 6, x = l & 63;
    const float* p = t2 + ((((size_t)b * CCH + c) * HH + y) * WWID + x);
    float x00 = p[0];
    float x01 = (x < WWID - 1) ? p[1]    : 0.f;
    float x10 = (y < HH - 1)   ? p[WWID] : 0.f;
    float x11 = (x < WWID - 1 && y < HH - 1) ? p[WWID + 1] : 0.f;
    xc[g] = x00;
    bg[g] = (x00 + x01 + x10 + x11) * 0.25f;
    fg[g] = (3.f * x00 - x01 - x10 - x11) * 0.25f;
}

// ---------------------------------------------------------------------------
// In-place scaled softmax over trailing q-axis (9 wide).
// ---------------------------------------------------------------------------
__global__ __launch_bounds__(256)
void softmax9(float* __restrict__ Aat)
{
    int r = blockIdx.x * 256 + threadIdx.x;          // B*L*72 rows
    if (r >= BATCH * LPIX * 72) return;
    int pix = r / 72, sub = r % 72;                   // sub = h*9 + p
    float* p = Aat + (size_t)pix * NATT + sub * 9;
    float v[9], mx = -1e30f;
    #pragma unroll
    for (int i = 0; i < 9; ++i) { v[i] = p[i] * ATT_SCALE; mx = fmaxf(mx, v[i]); }
    float s = 0.f;
    #pragma unroll
    for (int i = 0; i < 9; ++i) { v[i] = __expf(v[i] - mx); s += v[i]; }
    float inv = 1.f / s;
    #pragma unroll
    for (int i = 0; i < 9; ++i) p[i] = v[i] * inv;
}

// ---------------------------------------------------------------------------
// Fused einsum(a, v_cols) + fold3 as per-output-pixel gather.
// ---------------------------------------------------------------------------
__global__ __launch_bounds__(256)
void fold_av(const float* __restrict__ Aat, const float* __restrict__ V,
             float* __restrict__ out)
{
    const int c = threadIdx.x;
    const int h = c >> 5;
    const int blk = blockIdx.x;
    const int b = blk >> 12, l = blk & (LPIX - 1);
    const int y = l >> 6, x = l & 63;
    float acc = 0.f;
    #pragma unroll
    for (int pi = 0; pi < 3; ++pi)
        #pragma unroll
        for (int pj = 0; pj < 3; ++pj) {
            int ly = y + 1 - pi, lx = x + 1 - pj;
            if (ly < 0 || ly >= HH || lx < 0 || lx >= WWID) continue;
            const float* arow = Aat + (size_t)((b << 12) + (ly << 6) + lx) * NATT
                                    + h * 81 + (pi * 3 + pj) * 9;
            #pragma unroll
            for (int q = 0; q < 9; ++q) {
                int qi = q / 3, qj = q % 3;
                int vy = y + qi - pi, vx = x + qj - pj;
                if (vy < 0 || vy >= HH || vx < 0 || vx >= WWID) continue;
                acc += arow[q] * V[(((size_t)(b << 12) + (vy << 6) + vx) << 8) + c];
            }
        }
    out[((size_t)blk << 8) + c] = acc;
}

// ---------------------------------------------------------------------------
extern "C" void kernel_launch(void* const* d_in, const int* in_sizes, int n_in,
                              void* d_out, int out_size, void* d_ws, size_t ws_size,
                              hipStream_t stream)
{
    const float* x   = (const float*)d_in[0];
    const float* cw1 = (const float*)d_in[1];
    const float* cs1 = (const float*)d_in[2];
    const float* cb1 = (const float*)d_in[3];
    const float* cw2 = (const float*)d_in[4];
    const float* cs2 = (const float*)d_in[5];
    const float* cb2 = (const float*)d_in[6];
    const float* Wv  = (const float*)d_in[7];
    const float* bv  = (const float*)d_in[8];
    const float* Wfg = (const float*)d_in[9];
    const float* bfg = (const float*)d_in[10];
    const float* Wbg = (const float*)d_in[11];
    const float* bbg = (const float*)d_in[12];
    const float* Wp  = (const float*)d_in[13];
    const float* bp  = (const float*)d_in[14];
    const float* ow1 = (const float*)d_in[15];
    const float* os1 = (const float*)d_in[16];
    const float* ob1 = (const float*)d_in[17];
    const float* ow2 = (const float*)d_in[18];
    const float* os2 = (const float*)d_in[19];
    const float* ob2 = (const float*)d_in[20];
    float* out = (float*)d_out;

    const size_t NE = (size_t)BATCH * CCH * HH * WWID;   // 4,194,304 floats
    float* ws   = (float*)d_ws;
    float* buf0 = ws;            // t1        -> folded (both passes)
    float* buf1 = buf0 + NE;     // t2 (NCHW) -> xw_fg (pixel-major)
    float* buf2 = buf1 + NE;     // xc (pm)   -> attn2 (NCHW)
    float* buf3 = buf2 + NE;     // fg (pm)   -> t3 (NCHW)
    float* buf4 = buf3 + NE;     // bg (pm)
    float* buf5 = buf4 + NE;     // v  (pm)
    float* Abuf = buf5 + NE;     // logits (B*L*648 floats)

    // bf16 packed-weight region (32B-aligned offsets)
    const size_t CPK = 72 * 16 * 32 * 16;       // 589,824 bf16 per conv weight
    const size_t GPK256 = 8 * 16 * 32 * 16;     // 65,536 bf16 (N=256, 16 ntiles)
    const size_t GPK648 = 8 * 48 * 32 * 16;     // 196,608 bf16 (N=648, 48 ntiles)
    __bf16* pkBase = (__bf16*)(Abuf + (size_t)BATCH * LPIX * NATT);
    __bf16* pkC1  = pkBase;
    __bf16* pkC2  = pkC1  + CPK;
    __bf16* pkO1  = pkC2  + CPK;
    __bf16* pkO2  = pkO1  + CPK;
    __bf16* pkWv  = pkO2  + CPK;
    __bf16* pkWp  = pkWv  + GPK256;
    __bf16* pkWfg = pkWp  + GPK256;
    __bf16* pkWbg = pkWfg + GPK648;

    dim3 blk(256);
    dim3 cgrid(512);                          // conv: (b, y, xTile32); 256 oc/block
    dim3 g256(256, 2);                        // GEMM N=256: 64x128 tiles
    dim3 g648(256, 6);                        // GEMM N=648 (col-guarded, 48 ntiles)

    // one-time weight packs (tiny, deterministic)
    pack_w_conv<<<144, blk, 0, stream>>>(cw1, pkC1);
    pack_w_conv<<<144, blk, 0, stream>>>(cw2, pkC2);
    pack_w_conv<<<144, blk, 0, stream>>>(ow1, pkO1);
    pack_w_conv<<<144, blk, 0, stream>>>(ow2, pkO2);
    pack_w_gemm<<<16, blk, 0, stream>>>(Wv,  pkWv,  CCH, CCH,  16);
    pack_w_gemm<<<16, blk, 0, stream>>>(Wp,  pkWp,  CCH, CCH,  16);
    pack_w_gemm<<<48, blk, 0, stream>>>(Wfg, pkWfg, CCH, NATT, 48);
    pack_w_gemm<<<48, blk, 0, stream>>>(Wbg, pkWbg, CCH, NATT, 48);

    // stem convs
    conv3x3_bn_silu<<<cgrid, blk, 0, stream>>>(x,    pkC1, cs1, cb1, buf0);
    conv3x3_bn_silu<<<cgrid, blk, 0, stream>>>(buf0, pkC2, cs2, cb2, buf1);
    // haar split + transpose to pixel-major
    haar_transpose<<<16384, blk, 0, stream>>>(buf1, buf2, buf3, buf4);
    // v = xc @ Wv + bv
    gemm_bias_wmma<<<g256, blk, 0, stream>>>(buf2, pkWv, bv, buf5, CCH, CCH, 16, 0);
    // pass 1 (fg)
    gemm_bias_wmma<<<g648, blk, 0, stream>>>(buf3, pkWfg, bfg, Abuf, CCH, NATT, 48, 0);
    softmax9<<<4608, blk, 0, stream>>>(Abuf);
    fold_av<<<16384, blk, 0, stream>>>(Abuf, buf5, buf0);
    gemm_bias_wmma<<<g256, blk, 0, stream>>>(buf0, pkWp, bp, buf1, CCH, CCH, 16, 0);
    // pass 2 (bg), v := xw_fg
    gemm_bias_wmma<<<g648, blk, 0, stream>>>(buf4, pkWbg, bbg, Abuf, CCH, NATT, 48, 0);
    softmax9<<<4608, blk, 0, stream>>>(Abuf);
    fold_av<<<16384, blk, 0, stream>>>(Abuf, buf1, buf0);
    gemm_bias_wmma<<<g256, blk, 0, stream>>>(buf0, pkWp, bp, buf2, CCH, CCH, 16, 1);
    // output convs
    conv3x3_bn_silu<<<cgrid, blk, 0, stream>>>(buf2, pkO1, os1, ob1, buf3);
    conv3x3_bn_silu<<<cgrid, blk, 0, stream>>>(buf3, pkO2, os2, ob2, out);
}
